// GCN_26242250179008
// MI455X (gfx1250) — matrix operand order repaired and checked by reading, not compiled, not run on hardware
//
#include <hip/hip_runtime.h>
#include <hip/hip_bf16.h>

typedef __attribute__((ext_vector_type(16))) _Float16 v16h;
typedef __attribute__((ext_vector_type(8)))  _Float16 v8h;
typedef __attribute__((ext_vector_type(4)))  _Float16 v4h;
typedef __attribute__((ext_vector_type(8)))  float    v8f;
typedef __attribute__((ext_vector_type(4)))  float    v4f;

#define N_NODES   200000
#define FDIM      128
#define NCLASS    10
#define ROWP      136                    // padded f16 row stride (128+8) to stagger LDS banks
#define WAVES_PB  8
#define TM        16                     // node rows per wave tile
#define NTILES    (N_NODES / TM)         // 12500 (exact)
#define GRID1     ((NTILES + WAVES_PB - 1) / WAVES_PB)   // 1563

// ---- dynamic LDS layout (bytes) ----
#define W_BYTES    (128 * ROWP * 2)                       // 34816 per weight matrix (f16)
#define ACT_OFF    (3 * W_BYTES)                          // 104448
#define ACT_BYTES  (TM * ROWP * 2)                        // 4352 per wave
#define BIAS_OFF   (ACT_OFF + WAVES_PB * ACT_BYTES)       // 139264
#define WSUM_OFF   (BIAS_OFF + 3 * 128 * 4)               // 140800
#define SMEM_BYTES (WSUM_OFF + WAVES_PB * 2 * 128 * 4)    // 148992 (< 320KB WGP pool)

// A fragment (16x32 f16): lanes 0-15 -> row M=lane, K = k0+{0..7, 16..23}
//                         lanes 16-31 -> row M=lane-16, K = k0+{8..15, 24..31}
__device__ __forceinline__ v16h load_frag_A(const _Float16* base, int lane, int k0) {
    const int m   = lane & 15;
    const int sel = (lane >> 4) * 8;
    const _Float16* p = base + m * ROWP + k0 + sel;
    v8h lo = *(const v8h*)(p);           // ds_load_b128
    v8h hi = *(const v8h*)(p + 16);      // ds_load_b128
    return __builtin_shufflevector(lo, hi, 0,1,2,3,4,5,6,7,8,9,10,11,12,13,14,15);
}

// B fragment (32x16 f16), B[k,n] = W[n,k]: lanes 0-15 -> col N=lane, K = k0+{0..7, 16..23}
//                                          lanes 16-31 -> col N=lane-16, K = k0+{8..15, 24..31}
// Row-major W rows land directly in lanes -> no transpose needed.
__device__ __forceinline__ v16h load_frag_B(const _Float16* wmat, int lane, int nt, int k0) {
    const int n   = nt * 16 + (lane & 15);
    const int sel = (lane >> 4) * 8;
    const _Float16* p = wmat + n * ROWP + k0 + sel;
    v8h lo = *(const v8h*)(p);           // ds_load_b128
    v8h hi = *(const v8h*)(p + 16);      // ds_load_b128
    return __builtin_shufflevector(lo, hi, 0,1,2,3,4,5,6,7,8,9,10,11,12,13,14,15);
}

// One 16x128 = (16x128)@(128x128) layer: 8 N-tiles x 4 K-chunks = 32 WMMA
__device__ __forceinline__ void compute_layer(const _Float16* inact, const _Float16* wmat,
                                              v8f acc[8], int lane) {
    v16h a[4];
#pragma unroll
    for (int kc = 0; kc < 4; ++kc) a[kc] = load_frag_A(inact, lane, kc * 32);
#pragma unroll
    for (int nt = 0; nt < 8; ++nt) {
        v8f c = {};
#pragma unroll
        for (int kc = 0; kc < 4; ++kc) {
            v16h b = load_frag_B(wmat, lane, nt, kc * 32);
            c = __builtin_amdgcn_wmma_f32_16x16x32_f16(false, a[kc], false, b,
                                                       (short)0, c, false, false);
        }
        acc[nt] = c;
    }
}

// C/D layout: lanes 0-15: VGPR i -> (M=i, N=lane); lanes 16-31: (M=i+8, N=lane-16)
__device__ __forceinline__ void store_act(const v8f acc[8], _Float16* myact,
                                          const float* bvec, int lane, bool relu) {
    const int nlo   = lane & 15;
    const int mbase = (lane >> 4) * 8;
#pragma unroll
    for (int nt = 0; nt < 8; ++nt) {
        const int col = nt * 16 + nlo;
        const float b = bvec[col];
#pragma unroll
        for (int i = 0; i < 8; ++i) {
            float v = acc[nt][i] + b;
            if (relu) v = fmaxf(v, 0.f);
            myact[(mbase + i) * ROWP + col] = (_Float16)v;
        }
    }
}

__global__ __launch_bounds__(256, 1)
void gcn_fused_mlp(const float* __restrict__ x,
                   const float* __restrict__ W1, const float* __restrict__ b1,
                   const float* __restrict__ W2, const float* __restrict__ b2,
                   const float* __restrict__ W3, const float* __restrict__ b3,
                   float* __restrict__ partial)        // [GRID1][128]
{
    extern __shared__ char smem[];
    _Float16* wl    = (_Float16*)(smem);                 // [3][128*ROWP] f16 weights
    _Float16* actl  = (_Float16*)(smem + ACT_OFF);       // [8 waves][16*ROWP] f16 acts
    float*    biasl = (float*)(smem + BIAS_OFF);         // [3][128]
    float*    wsum  = (float*)(smem + WSUM_OFF);         // [8 waves][2][128]

    const int tid  = threadIdx.x;
    const int wave = tid >> 5;
    const int lane = tid & 31;

    // ---- stage weights f32 -> f16 (padded rows) and biases into LDS ----
    const float* Ws[3] = {W1, W2, W3};
#pragma unroll
    for (int m = 0; m < 3; ++m) {
        _Float16* dst = wl + m * (128 * ROWP);
        const float* src = Ws[m];
#pragma unroll
        for (int it = 0; it < 16; ++it) {
            const int idx = (it * 256 + tid) * 4;        // 4 f32, same row (128%4==0)
            const int r = idx >> 7, c = idx & 127;
            v4f v = *(const v4f*)(src + idx);            // global_load_b128
            v4h h = {(_Float16)v.x, (_Float16)v.y, (_Float16)v.z, (_Float16)v.w};
            *(v4h*)(dst + r * ROWP + c) = h;             // ds_store_b64
        }
    }
    if (tid < 128) {
        biasl[tid]       = b1[tid];
        biasl[128 + tid] = b2[tid];
        biasl[256 + tid] = b3[tid];
    }
#pragma unroll
    for (int i = 0; i < 8; ++i) wsum[i * 256 + tid] = 0.f;  // zero 8*2*128 slots
    __syncthreads();

    // ---- per-wave 16-node tile (wave-uniform guard: EXEC all-ones inside) ----
    const int  wtile  = blockIdx.x * WAVES_PB + wave;
    const bool active = (wtile < NTILES);
    _Float16*  myact  = actl + wave * (TM * ROWP);

    if (active) {
        const int m0 = wtile * TM;
        // load x tile (fp32, coalesced b128) -> f16 act tile
#pragma unroll
        for (int r = 0; r < TM; ++r) {
            v4f v = *(const v4f*)(x + (size_t)(m0 + r) * FDIM + lane * 4);
            v4h h = {(_Float16)v.x, (_Float16)v.y, (_Float16)v.z, (_Float16)v.w};
            *(v4h*)(myact + r * ROWP + lane * 4) = h;
        }

        v8f acc[8];
        // layer 1 + ReLU
        compute_layer(myact, wl, acc, lane);
        store_act(acc, myact, biasl, lane, true);
        // layer 2 + ReLU
        compute_layer(myact, wl + 128 * ROWP, acc, lane);
        store_act(acc, myact, biasl + 128, lane, true);
        // layer 3 (no ReLU) -> per-wave column sums for mean pool
        compute_layer(myact, wl + 2 * 128 * ROWP, acc, lane);
#pragma unroll
        for (int nt = 0; nt < 8; ++nt) {
            const int col = nt * 16 + (lane & 15);
            const float b = biasl[256 + col];
            float s = 0.f;
#pragma unroll
            for (int i = 0; i < 8; ++i) s += acc[nt][i] + b;
            wsum[wave * 256 + (lane >> 4) * 128 + col] = s;
        }
    }
    __syncthreads();

    // deterministic fixed-order block reduction -> one partial row per block
    if (tid < 128) {
        float s = 0.f;
#pragma unroll
        for (int i = 0; i < 16; ++i) s += wsum[i * 128 + tid];
        partial[(size_t)blockIdx.x * 128 + tid] = s;
    }
}

__global__ __launch_bounds__(128, 1)
void gcn_pool_head(const float* __restrict__ partial, int nblocks,
                   const float* __restrict__ Wl, const float* __restrict__ bl,
                   float* __restrict__ out)
{
    __shared__ float g[128];
    __shared__ float logits[NCLASS];
    const int tid = threadIdx.x;

    float s = 0.f;
    for (int b = 0; b < nblocks; ++b) s += partial[(size_t)b * 128 + tid];
    g[tid] = s * (1.0f / (float)N_NODES);
    __syncthreads();

    if (tid < NCLASS) {
        float acc = bl[tid];
#pragma unroll 8
        for (int k = 0; k < 128; ++k) acc += g[k] * Wl[tid * 128 + k];
        logits[tid] = acc;
    }
    __syncthreads();

    if (tid == 0) {
        float mx = logits[0];
        for (int i = 1; i < NCLASS; ++i) mx = fmaxf(mx, logits[i]);
        float se = 0.f;
        for (int i = 0; i < NCLASS; ++i) se += __expf(logits[i] - mx);
        const float lse = mx + __logf(se);
        for (int i = 0; i < NCLASS; ++i) out[i] = logits[i] - lse;
    }
}

extern "C" void kernel_launch(void* const* d_in, const int* in_sizes, int n_in,
                              void* d_out, int out_size, void* d_ws, size_t ws_size,
                              hipStream_t stream) {
    (void)in_sizes; (void)n_in; (void)out_size; (void)ws_size;
    const float* x  = (const float*)d_in[0];
    // d_in[1] = edge_index: dead with ChebConv K=1 (no propagation)
    const float* W1 = (const float*)d_in[2];
    const float* b1 = (const float*)d_in[3];
    const float* W2 = (const float*)d_in[4];
    const float* b2 = (const float*)d_in[5];
    const float* W3 = (const float*)d_in[6];
    const float* b3 = (const float*)d_in[7];
    const float* Wl = (const float*)d_in[8];
    const float* bl = (const float*)d_in[9];
    float* partial = (float*)d_ws;                 // GRID1*128 f32 = 800 KB scratch
    float* out     = (float*)d_out;

    static_assert(SMEM_BYTES <= 320 * 1024, "LDS over WGP budget");
    hipFuncSetAttribute(reinterpret_cast<const void*>(&gcn_fused_mlp),
                        hipFuncAttributeMaxDynamicSharedMemorySize, SMEM_BYTES);

    gcn_fused_mlp<<<GRID1, 256, SMEM_BYTES, stream>>>(x, W1, b1, W2, b2, W3, b3, partial);
    gcn_pool_head<<<1, 128, 0, stream>>>(partial, GRID1, Wl, bl, out);
}
